// PointNetSetAbstractionMsg_72533407695112
// MI455X (gfx1250) — matrix operand order repaired and checked by reading, not compiled, not run on hardware
//
#include <hip/hip_runtime.h>
#include <stdint.h>

// PointNet++ MSG set abstraction for MI455X (gfx1250, wave32, WMMA).
// Pipeline: FPS -> per-scale {ball query -> gather f16 -> [WMMA GEMM -> BN-stat
// reduce -> BN+ReLU (+max on last layer)] x3}. Activations kept in f16 to halve
// HBM traffic (23.3 TB/s chip; problem is bandwidth-bound, ~58 GFLOP total).
// GEMM: weights staged to LDS via CDNA5 async global->LDS copies (ASYNCcnt);
// GEMM is templated on (M-tiles, K-tiles) so B fragments are all loaded in one
// up-front clause and the k-loop fully unrolls (no rotation movs / branches).

#define NPT   512
#define NPTS  4096
#define NB    16
#define INCH  64
#define EPSV  1e-5f

typedef __attribute__((ext_vector_type(16))) _Float16 v16h;
typedef __attribute__((ext_vector_type(8)))  _Float16 v8h;
typedef __attribute__((ext_vector_type(8)))  float    v8f;

__device__ inline unsigned ballot32(bool p) {
#if __has_builtin(__builtin_amdgcn_ballot_w32)
  return __builtin_amdgcn_ballot_w32(p);
#else
  return (unsigned)__ballot(p);
#endif
}

// ---------------- Farthest point sampling: one block per batch ----------------
__global__ __launch_bounds__(256) void fps_kernel(const float* __restrict__ xyz,
                                                  int* __restrict__ fps_idx,
                                                  float* __restrict__ out_xyz) {
  __shared__ float dist[NPTS];
  __shared__ float rv[256];
  __shared__ int   ri[256];
  const int b = blockIdx.x, tid = threadIdx.x;
  const float* px = xyz + (size_t)b * 3 * NPTS;
  const float* py = px + NPTS;
  const float* pz = py + NPTS;
  for (int i = tid; i < NPTS; i += 256) dist[i] = 1e10f;
  __syncthreads();
  int far = 0;
  for (int it = 0; it < NPT; ++it) {
    if (tid == 0) {
      fps_idx[b * NPT + it] = far;
      out_xyz[(size_t)b * 3 * NPT + 0 * NPT + it] = px[far];
      out_xyz[(size_t)b * 3 * NPT + 1 * NPT + it] = py[far];
      out_xyz[(size_t)b * 3 * NPT + 2 * NPT + it] = pz[far];
    }
    const float cx = px[far], cy = py[far], cz = pz[far];
    float best = -1.f; int bi = NPTS - 1;
    for (int i = tid; i < NPTS; i += 256) {
      float dx = px[i] - cx, dy = py[i] - cy, dz = pz[i] - cz;
      float dd = fminf(dist[i], dx * dx + dy * dy + dz * dz);
      dist[i] = dd;
      if (dd > best) { best = dd; bi = i; }
    }
    rv[tid] = best; ri[tid] = bi;
    __syncthreads();
    for (int off = 128; off > 0; off >>= 1) {
      if (tid < off) {
        float o = rv[tid + off]; int oi = ri[tid + off];
        if (o > rv[tid] || (o == rv[tid] && oi < ri[tid])) { rv[tid] = o; ri[tid] = oi; }
      }
      __syncthreads();
    }
    far = ri[0];
    __syncthreads();
  }
}

// ------------- Ball query: one wave32 per query point (ascending idx) ---------
__global__ __launch_bounds__(256) void query_kernel(const float* __restrict__ xyz,
    const int* __restrict__ fps_idx, int* __restrict__ groups, int ns, float r2) {
  const int q = blockIdx.x * 8 + (threadIdx.x >> 5);
  const int lane = threadIdx.x & 31;
  const int b = q >> 9;
  const float* px = xyz + (size_t)b * 3 * NPTS;
  const float* py = px + NPTS;
  const float* pz = py + NPTS;
  const int fi = fps_idx[q];
  const float cx = px[fi], cy = py[fi], cz = pz[fi];
  int* g = groups + (size_t)q * ns;
  int count = 0, firstIdx = 0; bool have = false;
  for (int n0 = 0; n0 < NPTS && count < ns; n0 += 32) {
    int j = n0 + lane;
    float dx = px[j] - cx, dy = py[j] - cy, dz = pz[j] - cz;
    bool inb = (dx * dx + dy * dy + dz * dz) <= r2;
    unsigned mask = ballot32(inb);
    if (mask) {
      if (!have) { firstIdx = n0 + __builtin_ctz(mask); have = true; }
      int slot = count + __builtin_popcount(mask & ((1u << lane) - 1u));
      if (inb && slot < ns) g[slot] = j;
      count += __builtin_popcount(mask);
    }
  }
  for (int slot = count + lane; slot < ns; slot += 32) g[slot] = firstIdx;
}

// --- Gather grouped features into f16 X[s][96]: 64 pts-ch + 3 rel-xyz + pad ---
__global__ __launch_bounds__(256) void gather_kernel(const float* __restrict__ xyz,
    const float* __restrict__ points, const int* __restrict__ fps_idx,
    const int* __restrict__ groups, _Float16* __restrict__ X, int ns, long long total) {
  long long t = (long long)blockIdx.x * 256 + threadIdx.x;
  if (t >= total) return;
  int c = (int)(t % 96);
  long long s = t / 96;
  long long q = s / ns;
  int k = (int)(s % ns);
  int b = (int)(q >> 9);
  int gi = groups[q * ns + k];
  float v;
  if (c < 64) {
    v = points[((size_t)b * INCH + c) * NPTS + gi];
  } else if (c < 67) {
    int d = c - 64;
    int fi = fps_idx[q];
    const float* pd = xyz + ((size_t)b * 3 + d) * NPTS;
    v = pd[gi] - pd[fi];
  } else {
    v = 0.f;
  }
  X[s * 96 + c] = (_Float16)v;
}

// -------- Convert/pad weights (Cout x Cin f32) -> f16 [Cout][Kpad] ------------
__global__ void prep_weights_kernel(const float* __restrict__ w, _Float16* __restrict__ w16,
                                    int Cin, int Kpad, int total) {
  int t = blockIdx.x * 256 + threadIdx.x;
  if (t >= total) return;
  int kp = t % Kpad, m = t / Kpad;
  w16[t] = (_Float16)((kp < Cin) ? w[m * Cin + kp] : 0.f);
}

__global__ void zero_kernel(float* p, int n) {
  int t = blockIdx.x * 256 + threadIdx.x;
  if (t < n) p[t] = 0.f;
}

// ---------------- WMMA GEMM: Y[s][M] = W[M][K] * X[s][K] + bias ---------------
// Block = 256 threads = 8 waves; each wave owns 16 spatial columns and loops
// over all TM M-tiles. Weights staged once into LDS with async global->LDS
// copies (CDNA5 ASYNCcnt path). All KT B fragments are loaded in one clause
// up-front, then the fully-unrolled k/m loops run WMMAs against LDS A frags.
template<int TM, int KT>
__global__ __launch_bounds__(256) void gemm_kernel(const _Float16* __restrict__ X,
    const _Float16* __restrict__ W, const float* __restrict__ bias,
    _Float16* __restrict__ Y) {
  constexpr int M = TM * 16;
  constexpr int Kpad = KT * 32;
  constexpr int totalW = M * Kpad;
  __shared__ _Float16 sW[totalW];
  const int tid = threadIdx.x;
  // Async copy weights to LDS: per-lane 16B transfers, LDS addr in VGPR.
#pragma unroll
  for (int idx = tid * 8; idx < totalW; idx += 256 * 8) {
    unsigned lds_off = (unsigned)(size_t)&sW[idx];
    const _Float16* gp = &W[idx];
    asm volatile("global_load_async_to_lds_b128 %0, %1, off"
                 :: "v"(lds_off), "v"(gp) : "memory");
  }
  asm volatile("s_wait_asynccnt 0x0" ::: "memory");
  __syncthreads();

  const int wave = tid >> 5, lane = tid & 31;
  const int half = lane >> 4, l16 = lane & 15;
  const long long scol = (long long)blockIdx.x * 128 + wave * 16 + l16;
  const _Float16* xcol = X + scol * Kpad;

  // B frags: column = l16, chunks K[8h..8h+7] and K[16+8h..16+8h+7] per k-step.
  v16h bfrag[KT];
#pragma unroll
  for (int kt = 0; kt < KT; ++kt) {
    v8h b0 = *(const v8h*)(xcol + kt * 32 + half * 8);
    v8h b1 = *(const v8h*)(xcol + kt * 32 + half * 8 + 16);
    bfrag[kt] = __builtin_shufflevector(b0, b1, 0, 1, 2, 3, 4, 5, 6, 7,
                                        8, 9, 10, 11, 12, 13, 14, 15);
  }

  const v8f vz = {0.f, 0.f, 0.f, 0.f, 0.f, 0.f, 0.f, 0.f};
  v8f acc[TM];
#pragma unroll
  for (int mt = 0; mt < TM; ++mt) acc[mt] = vz;

#pragma unroll
  for (int kt = 0; kt < KT; ++kt) {
    const int kbase = kt * 32;
#pragma unroll
    for (int mt = 0; mt < TM; ++mt) {
      // A frag: row = mt*16+l16; chunks K[8h..8h+7] and K[16+8h..16+8h+7]
      const _Float16* arow = &sW[(size_t)(mt * 16 + l16) * Kpad + kbase + half * 8];
      v8h a0 = *(const v8h*)(arow);
      v8h a1 = *(const v8h*)(arow + 16);
      v16h afrag = __builtin_shufflevector(a0, a1, 0, 1, 2, 3, 4, 5, 6, 7,
                                           8, 9, 10, 11, 12, 13, 14, 15);
      acc[mt] = __builtin_amdgcn_wmma_f32_16x16x32_f16(
          false, afrag, false, bfrag[kt], (short)0, acc[mt], false, false);
    }
  }
#pragma unroll
  for (int mt = 0; mt < TM; ++mt) {
    const int mbase = mt * 16 + half * 8;  // 8 consecutive output rows per lane
    v8h hy;
#pragma unroll
    for (int e = 0; e < 8; ++e) hy[e] = (_Float16)(acc[mt][e] + bias[mbase + e]);
    *(v8h*)(Y + scol * M + mbase) = hy;
  }
}

// ------------- Per-channel sum / sumsq reduction (BN statistics) --------------
__global__ __launch_bounds__(256) void reduce_stats_kernel(const _Float16* __restrict__ Y,
    float* __restrict__ stats, long long total, int M) {
  __shared__ float ssum[128], ssq[128];
  int tid = threadIdx.x;
  for (int c = tid; c < M; c += 256) { ssum[c] = 0.f; ssq[c] = 0.f; }
  __syncthreads();
  long long stride = (long long)gridDim.x * 256;
  for (long long i = (long long)blockIdx.x * 256 + tid; i < total; i += stride) {
    float v = (float)Y[i];
    int c = (int)(i % M);
    atomicAdd(&ssum[c], v);
    atomicAdd(&ssq[c], v * v);
  }
  __syncthreads();
  for (int c = tid; c < M; c += 256) {
    atomicAdd(&stats[c], ssum[c]);
    atomicAdd(&stats[128 + c], ssq[c]);
  }
}

// ---------------- BN + ReLU -> f16 activations for next layer ----------------
__global__ __launch_bounds__(256) void bn_relu_kernel(const _Float16* __restrict__ Y,
    const float* __restrict__ stats, const float* __restrict__ g,
    const float* __restrict__ be, _Float16* __restrict__ Xo,
    long long total, int M, float invS) {
  long long i = (long long)blockIdx.x * 256 + threadIdx.x;
  if (i >= total) return;
  int c = (int)(i % M);
  float mu = stats[c] * invS;
  float var = stats[128 + c] * invS - mu * mu;
  float sc = rsqrtf(var + EPSV) * g[c];
  float v = ((float)Y[i] - mu) * sc + be[c];
  Xo[i] = (_Float16)fmaxf(v, 0.f);
}

// -------- BN + ReLU + max over nsample -> final f32 output channels ----------
__global__ __launch_bounds__(256) void bn_relu_max_kernel(const _Float16* __restrict__ Y,
    const float* __restrict__ stats, const float* __restrict__ g,
    const float* __restrict__ be, float* __restrict__ out,
    int M, int ns, int coff, float invS) {
  long long t = (long long)blockIdx.x * 256 + threadIdx.x;
  long long total = (long long)NB * NPT * M;
  if (t >= total) return;
  int c = (int)(t % M);
  long long q = t / M;
  int b = (int)(q >> 9), p = (int)(q & 511);
  float mu = stats[c] * invS;
  float var = stats[128 + c] * invS - mu * mu;
  float sc = rsqrtf(var + EPSV) * g[c];
  float sh = be[c] - mu * sc;
  const _Float16* yq = Y + (q * (long long)ns) * M + c;
  float m = -1e30f;
  for (int k = 0; k < ns; ++k) m = fmaxf(m, (float)yq[(long long)k * M] * sc + sh);
  m = fmaxf(m, 0.f);
  out[(size_t)24576 + ((size_t)b * 320 + coff + c) * NPT + p] = m;
}

static void launch_gemm(int TM, int KT, unsigned blocks, hipStream_t stream,
                        const _Float16* X, const _Float16* W, const float* bias,
                        _Float16* Y) {
#define GEMM_CASE(tm, kt) \
  if (TM == tm && KT == kt) { \
    gemm_kernel<tm, kt><<<blocks, 256, 0, stream>>>(X, W, bias, Y); return; }
  GEMM_CASE(2, 1) GEMM_CASE(2, 2) GEMM_CASE(2, 3)
  GEMM_CASE(4, 1) GEMM_CASE(4, 2) GEMM_CASE(4, 3)
  GEMM_CASE(6, 1) GEMM_CASE(6, 2) GEMM_CASE(6, 3)
  GEMM_CASE(8, 1) GEMM_CASE(8, 2) GEMM_CASE(8, 3)
#undef GEMM_CASE
}

extern "C" void kernel_launch(void* const* d_in, const int* in_sizes, int n_in,
                              void* d_out, int out_size, void* d_ws, size_t ws_size,
                              hipStream_t stream) {
  const float* xyz    = (const float*)d_in[0];
  const float* points = (const float*)d_in[1];
  static const int   MLP[3][3] = {{32, 32, 64}, {64, 64, 128}, {64, 96, 128}};
  static const int   NSAMP[3]  = {16, 32, 128};
  static const float RAD2[3]   = {0.01f, 0.04f, 0.16f};
  static const int   COFF[3]   = {0, 64, 192};

  char* ws = (char*)d_ws;
  int*      fps    = (int*)ws;                                        // 32 KB
  int*      groups = (int*)(ws + (1 << 16));                          // <= 4 MB
  float*    stats  = (float*)(ws + (1 << 16) + (4 << 20));            // 1 KB
  _Float16* w16    = (_Float16*)(ws + (1 << 16) + (4 << 20) + 4096);  // <= 24 KB
  const size_t big   = (size_t)8 << 20;
  const size_t BIGSZ = (size_t)NB * NPT * 128 * 128 * sizeof(_Float16); // 256 MB
  _Float16* XA = (_Float16*)(ws + big);
  _Float16* XB = (_Float16*)(ws + big + BIGSZ);
  _Float16* Yb = (_Float16*)(ws + big + 2 * BIGSZ);
  float* out = (float*)d_out;

  fps_kernel<<<NB, 256, 0, stream>>>(xyz, fps, out);

  int pi = 2;
  for (int i = 0; i < 3; ++i) {
    int ns = NSAMP[i];
    long long S = (long long)NB * NPT * ns;
    query_kernel<<<(NB * NPT) / 8, 256, 0, stream>>>(xyz, fps, groups, ns, RAD2[i]);
    long long gtot = S * 96;
    gather_kernel<<<(unsigned)((gtot + 255) / 256), 256, 0, stream>>>(
        xyz, points, fps, groups, XA, ns, gtot);

    const _Float16* cur = XA;
    _Float16* nxt = XB;
    int Cin = INCH + 3;
    for (int j = 0; j < 3; ++j) {
      int M = MLP[i][j];
      int Kpad = (j == 0) ? 96 : Cin;  // 67 -> 96; later Cin already %32==0
      const float* w  = (const float*)d_in[pi++];
      const float* bb = (const float*)d_in[pi++];
      const float* gg = (const float*)d_in[pi++];
      const float* be = (const float*)d_in[pi++];
      int wtot = M * Kpad;
      prep_weights_kernel<<<(wtot + 255) / 256, 256, 0, stream>>>(w, w16, Cin, Kpad, wtot);

      unsigned gblocks = (unsigned)(S / 128);  // always exact (powers of two)
      launch_gemm(M / 16, Kpad / 32, gblocks, stream, cur, w16, bb, Yb);

      zero_kernel<<<1, 256, 0, stream>>>(stats, 256);
      long long ytot = S * M;
      long long rb = (ytot + 256LL * 64 - 1) / (256LL * 64);
      unsigned rblocks = rb < 1 ? 1u : (unsigned)rb;
      reduce_stats_kernel<<<rblocks, 256, 0, stream>>>(Yb, stats, ytot, M);

      float invS = 1.0f / (float)S;
      if (j < 2) {
        bn_relu_kernel<<<(unsigned)((ytot + 255) / 256), 256, 0, stream>>>(
            Yb, stats, gg, be, nxt, ytot, M, invS);
        const _Float16* t = cur; cur = nxt; nxt = (_Float16*)t;
      } else {
        long long mtot = (long long)NB * NPT * M;
        bn_relu_max_kernel<<<(unsigned)((mtot + 255) / 256), 256, 0, stream>>>(
            Yb, stats, gg, be, out, M, ns, COFF[i], invS);
      }
      Cin = M;
    }
  }
  (void)in_sizes; (void)n_in; (void)out_size; (void)ws_size;
}
